// Block_40364102648000
// MI455X (gfx1250) — compile-verified, matrix operand order checked by reading, not run on hardware
//
#include <hip/hip_runtime.h>

#define NODES 50000
#define EDGES 800000
#define DIM 64
#define BN_EPS 1e-5f

static_assert(NODES % 16 == 0, "row tiles must be exact");

typedef __attribute__((ext_vector_type(16))) _Float16 v16h;
typedef __attribute__((ext_vector_type(8)))  _Float16 v8h;
typedef __attribute__((ext_vector_type(8)))  float    v8f;

union v16h_u {
    v16h full;
    v8h  part[2];
};

// Per ISA 7.12.2, for 16-bit A (and mirrored B) operands in wave32:
// element j of the v16h holds K = j + hi*8        for j = 0..7
//                            K = 16 + (j-8) + hi*8 for j = 8..15
// i.e. two contiguous 8-half (16-byte) runs per lane -> two ds_load_b128.
__device__ __forceinline__ v16h load_opnd16(const _Float16* base /*row of 16-bit K-major data*/,
                                            int kb, int hi) {
    v16h_u u;
    u.part[0] = *(const v8h*)(base + kb + hi * 8);
    u.part[1] = *(const v8h*)(base + kb + 16 + hi * 8);
    return u.full;
}

// ---------------------------------------------------------------------------
// agg = src (simple float4 copy)
// ---------------------------------------------------------------------------
__global__ void copy_f32x4(const float4* __restrict__ src, float4* __restrict__ dst, int n4) {
    int i = blockIdx.x * blockDim.x + threadIdx.x;
    if (i < n4) dst[i] = src[i];
}

// ---------------------------------------------------------------------------
// agg[dst[e]] += h[src[e]]  (edge-parallel, float4 gather + 4 f32 atomics)
// ---------------------------------------------------------------------------
__global__ void scatter_add_edges(const float* __restrict__ h,
                                  const int* __restrict__ src,
                                  const int* __restrict__ dst,
                                  float* __restrict__ agg) {
    int tid = blockIdx.x * blockDim.x + threadIdx.x;
    int e = tid >> 4;
    if (e >= EDGES) return;
    int q = tid & 15;
    int s = src[e];
    int d = dst[e];
    const float4 v = *(const float4*)(h + s * DIM + q * 4);
    float* p = agg + d * DIM + q * 4;
    atomicAdd(p + 0, v.x);
    atomicAdd(p + 1, v.y);
    atomicAdd(p + 2, v.z);
    atomicAdd(p + 3, v.w);
}

// ---------------------------------------------------------------------------
// Out(N x 64) = A(N x 64) @ W(64 x 64) + bias   -- f16 WMMA, f32 accumulate
// block = 128 threads (4 waves); block owns 16 rows, wave owns 16 cols.
// A staged row-major (K contiguous); W staged TRANSPOSED (Wt[n][k]) so both
// operands are two 16B LDS loads per lane.
// ---------------------------------------------------------------------------
__global__ __launch_bounds__(128) void gemm64_bias_wmma(
    const float* __restrict__ A, const float* __restrict__ W,
    const float* __restrict__ bias, float* __restrict__ Out) {
    __shared__ __align__(16) _Float16 As[16][DIM];       // [m][k]
    __shared__ __align__(16) _Float16 Wt[DIM][DIM];      // [n][k]
    int rowBase = blockIdx.x * 16;

    for (int idx = threadIdx.x; idx < 16 * DIM; idx += 128) {
        int r = idx >> 6, c = idx & 63;
        As[r][c] = (_Float16)A[(rowBase + r) * DIM + c];
    }
    for (int idx = threadIdx.x; idx < DIM * DIM; idx += 128) {
        int k = idx >> 6, n = idx & 63;
        Wt[n][k] = (_Float16)W[idx];
    }
    __syncthreads();

    int lane   = threadIdx.x & 31;
    int wave   = threadIdx.x >> 5;   // column tile 0..3
    int laneLo = lane & 15;
    int hi     = lane >> 4;
    int col    = wave * 16 + laneLo;

    float bv = bias[col];

    v8f acc = {};
#pragma unroll
    for (int kb = 0; kb < DIM; kb += 32) {
        v16h a = load_opnd16(&As[laneLo][0], kb, hi);
        v16h b = load_opnd16(&Wt[col][0],    kb, hi);
        acc = __builtin_amdgcn_wmma_f32_16x16x32_f16(
            false, a, false, b, (short)0, acc, false, false);
    }

#pragma unroll
    for (int i = 0; i < 8; ++i) {
        int r = rowBase + i + hi * 8;
        Out[r * DIM + col] = acc[i] + bv;
    }
}

// ---------------------------------------------------------------------------
// Out(N x 64) = H1 @ W[0:64,:] + H2 @ W[64:128,:] + bias   (JK cat + linear)
// ---------------------------------------------------------------------------
__global__ __launch_bounds__(128) void jk_linear_wmma(
    const float* __restrict__ H1, const float* __restrict__ H2,
    const float* __restrict__ W /*128x64 (k-major)*/, const float* __restrict__ bias,
    float* __restrict__ Out) {
    __shared__ __align__(16) _Float16 As[2][16][DIM];    // [part][m][k]
    __shared__ __align__(16) _Float16 Wt[DIM][2 * DIM];  // [n][kk]
    int rowBase = blockIdx.x * 16;

    for (int idx = threadIdx.x; idx < 16 * DIM; idx += 128) {
        int r = idx >> 6, c = idx & 63;
        As[0][r][c] = (_Float16)H1[(rowBase + r) * DIM + c];
        As[1][r][c] = (_Float16)H2[(rowBase + r) * DIM + c];
    }
    for (int idx = threadIdx.x; idx < 2 * DIM * DIM; idx += 128) {
        int kk = idx >> 6, n = idx & 63;
        Wt[n][kk] = (_Float16)W[idx];
    }
    __syncthreads();

    int lane   = threadIdx.x & 31;
    int wave   = threadIdx.x >> 5;
    int laneLo = lane & 15;
    int hi     = lane >> 4;
    int col    = wave * 16 + laneLo;

    float bv = bias[col];

    v8f acc = {};
#pragma unroll
    for (int part = 0; part < 2; ++part) {
#pragma unroll
        for (int kb = 0; kb < DIM; kb += 32) {
            v16h a = load_opnd16(&As[part][laneLo][0], kb, hi);
            v16h b = load_opnd16(&Wt[col][0], part * DIM + kb, hi);
            acc = __builtin_amdgcn_wmma_f32_16x16x32_f16(
                false, a, false, b, (short)0, acc, false, false);
        }
    }

#pragma unroll
    for (int i = 0; i < 8; ++i) {
        int r = rowBase + i + hi * 8;
        Out[r * DIM + col] = acc[i] + bv;
    }
}

// ---------------------------------------------------------------------------
// BN stats: stats[c] += sum, stats[64+c] += sumsq (over the node axis)
// ---------------------------------------------------------------------------
__global__ void zero_f32(float* __restrict__ p, int n) {
    int i = blockIdx.x * blockDim.x + threadIdx.x;
    if (i < n) p[i] = 0.0f;
}

__global__ __launch_bounds__(256) void bn_stats(
    const float* __restrict__ X, float* __restrict__ stats, int nRows) {
    __shared__ float ssum[4][DIM];
    __shared__ float ssq[4][DIM];
    int c = threadIdx.x & 63;
    int g = threadIdx.x >> 6;  // 0..3
    float sum = 0.0f, sq = 0.0f;
    for (int r = blockIdx.x * 4 + g; r < nRows; r += gridDim.x * 4) {
        float v = X[r * DIM + c];
        sum += v;
        sq  += v * v;
    }
    ssum[g][c] = sum;
    ssq[g][c]  = sq;
    __syncthreads();
    if (threadIdx.x < DIM) {
        float s = ssum[0][c] + ssum[1][c] + ssum[2][c] + ssum[3][c];
        float q = ssq[0][c]  + ssq[1][c]  + ssq[2][c]  + ssq[3][c];
        atomicAdd(&stats[c], s);
        atomicAdd(&stats[DIM + c], q);
    }
}

// ---------------------------------------------------------------------------
// X = relu((X - mean) * rsqrt(var + eps) * gamma + beta), in place
// ---------------------------------------------------------------------------
__global__ void bn_relu_apply(float* __restrict__ X, const float* __restrict__ stats,
                              const float* __restrict__ gamma,
                              const float* __restrict__ beta, int nRows) {
    int idx = blockIdx.x * blockDim.x + threadIdx.x;
    if (idx >= nRows * DIM) return;
    int c = idx & 63;
    float invN = 1.0f / (float)nRows;
    float m    = stats[c] * invN;
    float var  = stats[DIM + c] * invN - m * m;
    float v = (X[idx] - m) * rsqrtf(var + BN_EPS) * gamma[c] + beta[c];
    X[idx] = v > 0.0f ? v : 0.0f;
}

// ---------------------------------------------------------------------------
extern "C" void kernel_launch(void* const* d_in, const int* in_sizes, int n_in,
                              void* d_out, int out_size, void* d_ws, size_t ws_size,
                              hipStream_t stream) {
    (void)in_sizes; (void)n_in; (void)out_size; (void)ws_size;

    const float* x   = (const float*)d_in[0];
    const int*   ei  = (const int*)d_in[1];
    const int*   src = ei;
    const int*   dst = ei + EDGES;
    const float* w0a = (const float*)d_in[2];
    const float* b0a = (const float*)d_in[3];
    const float* g0a = (const float*)d_in[4];
    const float* e0a = (const float*)d_in[5];
    const float* w0b = (const float*)d_in[6];
    const float* b0b = (const float*)d_in[7];
    const float* g0b = (const float*)d_in[8];
    const float* e0b = (const float*)d_in[9];
    const float* w1a = (const float*)d_in[10];
    const float* b1a = (const float*)d_in[11];
    const float* g1a = (const float*)d_in[12];
    const float* e1a = (const float*)d_in[13];
    const float* w1b = (const float*)d_in[14];
    const float* b1b = (const float*)d_in[15];
    const float* g1b = (const float*)d_in[16];
    const float* e1b = (const float*)d_in[17];
    const float* lw  = (const float*)d_in[18];
    const float* lb  = (const float*)d_in[19];
    float* out = (float*)d_out;

    // Workspace: A (agg / h2), B (intermediate), C (h1), stats (128 f32)
    float* A = (float*)d_ws;
    float* B = A + (size_t)NODES * DIM;
    float* C = B + (size_t)NODES * DIM;
    float* stats = C + (size_t)NODES * DIM;

    const int n4      = NODES * DIM / 4;
    const int cpGrid  = (n4 + 255) / 256;
    const int edGrid  = (EDGES * 16 + 255) / 256;
    const int gmGrid  = NODES / 16;
    const int elGrid  = (NODES * DIM + 255) / 256;

    // ---- GIN layer 0 ----
    copy_f32x4<<<cpGrid, 256, 0, stream>>>((const float4*)x, (float4*)A, n4);
    scatter_add_edges<<<edGrid, 256, 0, stream>>>(x, src, dst, A);

    gemm64_bias_wmma<<<gmGrid, 128, 0, stream>>>(A, w0a, b0a, B);
    zero_f32<<<1, 128, 0, stream>>>(stats, 2 * DIM);
    bn_stats<<<256, 256, 0, stream>>>(B, stats, NODES);
    bn_relu_apply<<<elGrid, 256, 0, stream>>>(B, stats, g0a, e0a, NODES);

    gemm64_bias_wmma<<<gmGrid, 128, 0, stream>>>(B, w0b, b0b, C);
    zero_f32<<<1, 128, 0, stream>>>(stats, 2 * DIM);
    bn_stats<<<256, 256, 0, stream>>>(C, stats, NODES);
    bn_relu_apply<<<elGrid, 256, 0, stream>>>(C, stats, g0b, e0b, NODES);   // C = h1

    // ---- GIN layer 1 ----
    copy_f32x4<<<cpGrid, 256, 0, stream>>>((const float4*)C, (float4*)A, n4);
    scatter_add_edges<<<edGrid, 256, 0, stream>>>(C, src, dst, A);

    gemm64_bias_wmma<<<gmGrid, 128, 0, stream>>>(A, w1a, b1a, B);
    zero_f32<<<1, 128, 0, stream>>>(stats, 2 * DIM);
    bn_stats<<<256, 256, 0, stream>>>(B, stats, NODES);
    bn_relu_apply<<<elGrid, 256, 0, stream>>>(B, stats, g1a, e1a, NODES);

    gemm64_bias_wmma<<<gmGrid, 128, 0, stream>>>(B, w1b, b1b, A);
    zero_f32<<<1, 128, 0, stream>>>(stats, 2 * DIM);
    bn_stats<<<256, 256, 0, stream>>>(A, stats, NODES);
    bn_relu_apply<<<elGrid, 256, 0, stream>>>(A, stats, g1b, e1b, NODES);   // A = h2

    // ---- JK cat + final linear ----
    jk_linear_wmma<<<gmGrid, 128, 0, stream>>>(C, A, lw, lb, out);
}
